// LstmCellWithProjection_57990648431130
// MI455X (gfx1250) — matrix operand order repaired and checked by reading, not compiled, not run on hardware
//
#include <hip/hip_runtime.h>
#include <hip/hip_bf16.h>

// ---------------------------------------------------------------------------
// LSTM cell with projection on gfx1250 (MI455X).
// bf16 WMMA (v_wmma_f32_16x16x32_bf16) with f32 accumulation.
// Per timestep: kernel A = fused input+recurrent gates GEMM + cell update,
//               kernel B = split-K projection GEMM + clip.
// ---------------------------------------------------------------------------

typedef __bf16 bf16_t;
typedef __attribute__((ext_vector_type(16))) __bf16 v16bf;
typedef __attribute__((ext_vector_type(8)))  __bf16 v8bf;
typedef __attribute__((ext_vector_type(8)))  float  v8f;

#define CELL   4096
#define HID    512
#define DIM    512
#define BATCH  32
#define TSTEPS 128
#define MEM_CLIP  3.0f
#define PROJ_CLIP 3.0f

__device__ __forceinline__ v8f wmma_bf16(v16bf a, v16bf b, v8f c) {
  // 8 args: (neg_a, A, neg_b, B, c_mod, C, reuse_a, reuse_b)
  return __builtin_amdgcn_wmma_f32_16x16x32_bf16(false, a, false, b, (short)0, c,
                                                 false, false);
}

// A fragment (16x32 bf16, M = row per lane): lane holds 16 consecutive K values
// starting at k0 + 16*(lane>>4).
__device__ __forceinline__ v16bf load_a_frag(const bf16_t* __restrict__ base,
                                             size_t row_stride, int m, int k0,
                                             int lane) {
  const int half = lane >> 4;
  const bf16_t* p = base + (size_t)m * row_stride + (size_t)k0;
  v8bf lo = *(const v8bf*)(p + 16 * half);
  v8bf hi = *(const v8bf*)(p + 16 * half + 8);
  v16bf r;
#pragma unroll
  for (int i = 0; i < 8; ++i) { r[i] = lo[i]; r[i + 8] = hi[i]; }
  return r;
}

// B fragment (32x16 bf16, N = lane&15): lane holds 16 consecutive K values
// starting at k0 + 16*(lane>>4).  wrow = &W[n][0] (row-major [N][K]).
__device__ __forceinline__ v16bf load_b_frag(const bf16_t* __restrict__ wrow,
                                             int k0, int lane) {
  const int half = lane >> 4;
  return *(const v16bf*)(wrow + k0 + 16 * half);
}

__device__ __forceinline__ float sigmoidf_(float x) {
  return 1.0f / (1.0f + __expf(-x));
}
__device__ __forceinline__ float clipf_(float x, float c) {
  return fminf(c, fmaxf(-c, x));
}

// ---------------------------------------------------------------------------
// Kernel A: gates = x_t @ W_in^T + h @ W_state^T + b ; cell update ; pre.
// Grid: 32 blocks x 256 threads.  Wave owns 16 cells, computes all 4 gate
// tiles (i/f/m/o) for M=32 (two 16-row tiles) -> cell update in registers.
// ---------------------------------------------------------------------------
__global__ __launch_bounds__(256) void lstm_gates_cell_kernel(
    const bf16_t* __restrict__ x_bf,    // [B, T, D] bf16
    const bf16_t* __restrict__ Win_bf,  // [4*CELL, D] bf16 (row-major)
    const bf16_t* __restrict__ Wst_bf,  // [4*CELL, HID] bf16
    const float*  __restrict__ b_state, // [4*CELL]
    const bf16_t* __restrict__ h_bf,    // [B, HID] bf16
    float*        __restrict__ c,       // [B, CELL] f32 state
    bf16_t*       __restrict__ pre_bf,  // [B, CELL] bf16 (o * tanh(c_new))
    int t) {
  const int lane = threadIdx.x & 31;
  const int wave = threadIdx.x >> 5;
  const int ncell = blockIdx.x * 128 + wave * 16 + (lane & 15);
  const int half = lane >> 4;

  v8f acc[4][2];
#pragma unroll
  for (int g = 0; g < 4; ++g)
#pragma unroll
    for (int mt = 0; mt < 2; ++mt) acc[g][mt] = (v8f){0,0,0,0,0,0,0,0};

  // ---- Part 1: input projection, K over D=512 -----------------------------
  const bf16_t* xbase = x_bf + (size_t)t * DIM;  // element (m,k) at m*T*D + t*D + k
#pragma unroll 4
  for (int s = 0; s < DIM / 32; ++s) {
    const int k0 = s * 32;
    v16bf a0 = load_a_frag(xbase, (size_t)TSTEPS * DIM, (lane & 15), k0, lane);
    v16bf a1 = load_a_frag(xbase, (size_t)TSTEPS * DIM, (lane & 15) + 16, k0, lane);
#pragma unroll
    for (int g = 0; g < 4; ++g) {
      const bf16_t* wrow = Win_bf + ((size_t)g * CELL + ncell) * DIM;
      v16bf bf = load_b_frag(wrow, k0, lane);
      acc[g][0] = wmma_bf16(a0, bf, acc[g][0]);
      acc[g][1] = wmma_bf16(a1, bf, acc[g][1]);
    }
  }

  // ---- Part 2: recurrent, K over HID=512 ----------------------------------
#pragma unroll 4
  for (int s = 0; s < HID / 32; ++s) {
    const int k0 = s * 32;
    v16bf a0 = load_a_frag(h_bf, HID, (lane & 15), k0, lane);
    v16bf a1 = load_a_frag(h_bf, HID, (lane & 15) + 16, k0, lane);
#pragma unroll
    for (int g = 0; g < 4; ++g) {
      const bf16_t* wrow = Wst_bf + ((size_t)g * CELL + ncell) * HID;
      v16bf bf = load_b_frag(wrow, k0, lane);
      acc[g][0] = wmma_bf16(a0, bf, acc[g][0]);
      acc[g][1] = wmma_bf16(a1, bf, acc[g][1]);
    }
  }

  // ---- Cell update (registers) --------------------------------------------
  const float bias_i = b_state[0 * CELL + ncell];
  const float bias_f = b_state[1 * CELL + ncell];
  const float bias_m = b_state[2 * CELL + ncell];
  const float bias_o = b_state[3 * CELL + ncell];

#pragma unroll
  for (int mt = 0; mt < 2; ++mt) {
#pragma unroll
    for (int r = 0; r < 8; ++r) {
      const int b = mt * 16 + half * 8 + r;     // batch row of this C element
      float ig = acc[0][mt][r] + bias_i;
      float fg = acc[1][mt][r] + bias_f;
      float mi = acc[2][mt][r] + bias_m;
      float og = acc[3][mt][r] + bias_o;
      float iv = sigmoidf_(ig);
      float fv = sigmoidf_(fg);
      float mv = tanhf(mi);
      float ov = sigmoidf_(og);
      const size_t ci = (size_t)b * CELL + ncell;
      float cn = clipf_(iv * mv + fv * c[ci], MEM_CLIP);
      c[ci] = cn;
      pre_bf[ci] = (bf16_t)(ov * tanhf(cn));
    }
  }
}

// ---------------------------------------------------------------------------
// Kernel B: h_new = clip(pre @ W_proj^T).  M=32, N=512, K=4096.
// Grid: 32 blocks (one 16-col N tile each) x 8 waves; split-K across waves
// (512 K each), LDS reduction, then clip + write output + bf16 h.
// ---------------------------------------------------------------------------
__global__ __launch_bounds__(256) void lstm_proj_kernel(
    const bf16_t* __restrict__ pre_bf,  // [B, CELL] bf16
    const bf16_t* __restrict__ Wp_bf,   // [HID, CELL] bf16
    float*        __restrict__ out,     // [B, T, HID] f32 (d_out)
    bf16_t*       __restrict__ h_bf,    // [B, HID] bf16 (next-step A)
    int t) {
  __shared__ float lds[8 * 2 * 8 * 32];  // [wave][mt][r][lane] = 16 KB
  const int lane = threadIdx.x & 31;
  const int wave = threadIdx.x >> 5;
  const int ntile = blockIdx.x;  // 0..31
  const int ncol = ntile * 16 + (lane & 15);
  const bf16_t* wrow = Wp_bf + (size_t)ncol * CELL;

  v8f acc0 = (v8f){0,0,0,0,0,0,0,0};
  v8f acc1 = (v8f){0,0,0,0,0,0,0,0};
  const int kbase = wave * (CELL / 8);  // 512 K per wave
#pragma unroll 4
  for (int s = 0; s < (CELL / 8) / 32; ++s) {
    const int k0 = kbase + s * 32;
    v16bf a0 = load_a_frag(pre_bf, CELL, (lane & 15), k0, lane);
    v16bf a1 = load_a_frag(pre_bf, CELL, (lane & 15) + 16, k0, lane);
    v16bf bf = load_b_frag(wrow, k0, lane);
    acc0 = wmma_bf16(a0, bf, acc0);
    acc1 = wmma_bf16(a1, bf, acc1);
  }
#pragma unroll
  for (int r = 0; r < 8; ++r) {
    lds[((wave * 2 + 0) * 8 + r) * 32 + lane] = acc0[r];
    lds[((wave * 2 + 1) * 8 + r) * 32 + lane] = acc1[r];
  }
  __syncthreads();

  // 512 output elements, 256 threads -> 2 each.
  for (int q = threadIdx.x; q < 512; q += 256) {
    const int mt = q >> 8;
    const int r = (q >> 5) & 7;
    const int l = q & 31;
    float s = 0.0f;
#pragma unroll
    for (int w = 0; w < 8; ++w) s += lds[((w * 2 + mt) * 8 + r) * 32 + l];
    s = clipf_(s, PROJ_CLIP);
    const int b = mt * 16 + (l >> 4) * 8 + r;
    const int n = ntile * 16 + (l & 15);
    out[((size_t)b * TSTEPS + t) * HID + n] = s;
    h_bf[b * HID + n] = (bf16_t)s;
  }
}

// ---------------------------------------------------------------------------
// Helpers: fp32 -> bf16 conversion, state init, final-state epilogue.
// ---------------------------------------------------------------------------
__global__ void f32_to_bf16_kernel(const float* __restrict__ in,
                                   bf16_t* __restrict__ out, size_t n) {
  size_t i = (size_t)blockIdx.x * blockDim.x + threadIdx.x;
  if (i < n) out[i] = (bf16_t)in[i];
}

__global__ void init_state_kernel(bf16_t* __restrict__ h_bf,
                                  float* __restrict__ c) {
  int i = blockIdx.x * blockDim.x + threadIdx.x;
  if (i < BATCH * HID) h_bf[i] = (bf16_t)0.0f;
  if (i < BATCH * CELL) c[i] = 0.0f;
}

// d_out layout: outputs[32,128,512] | hT[1,32,512] | cT[1,32,4096]
__global__ void epilogue_kernel(const float* __restrict__ out,
                                const float* __restrict__ c,
                                float* __restrict__ dst) {
  int i = blockIdx.x * blockDim.x + threadIdx.x;
  const size_t OUT_N = (size_t)BATCH * TSTEPS * HID;  // 2097152
  if (i < BATCH * HID) {
    int b = i >> 9, n = i & 511;
    dst[OUT_N + i] = out[((size_t)b * TSTEPS + (TSTEPS - 1)) * HID + n];
  }
  if (i < BATCH * CELL) {
    dst[OUT_N + BATCH * HID + i] = c[i];
  }
}

// ---------------------------------------------------------------------------
extern "C" void kernel_launch(void* const* d_in, const int* in_sizes, int n_in,
                              void* d_out, int out_size, void* d_ws,
                              size_t ws_size, hipStream_t stream) {
  const float* x       = (const float*)d_in[0];  // [32,128,512]
  const float* W_in    = (const float*)d_in[1];  // [16384,512]
  const float* W_state = (const float*)d_in[2];  // [16384,512]
  const float* b_state = (const float*)d_in[3];  // [16384]
  const float* W_proj  = (const float*)d_in[4];  // [512,4096]
  float* out = (float*)d_out;

  // Workspace carve-up (256B aligned).
  auto align_up = [](size_t v) { return (v + 255) & ~(size_t)255; };
  char* p = (char*)d_ws;
  const size_t nx  = (size_t)BATCH * TSTEPS * DIM;   // 2,097,152
  const size_t nwi = (size_t)4 * CELL * DIM;         // 8,388,608
  const size_t nws = (size_t)4 * CELL * HID;         // 8,388,608
  const size_t nwp = (size_t)HID * CELL;             // 2,097,152
  bf16_t* x_bf   = (bf16_t*)p; p += align_up(nx * sizeof(bf16_t));
  bf16_t* Win_bf = (bf16_t*)p; p += align_up(nwi * sizeof(bf16_t));
  bf16_t* Wst_bf = (bf16_t*)p; p += align_up(nws * sizeof(bf16_t));
  bf16_t* Wp_bf  = (bf16_t*)p; p += align_up(nwp * sizeof(bf16_t));
  bf16_t* h_bf   = (bf16_t*)p; p += align_up((size_t)BATCH * HID * sizeof(bf16_t));
  bf16_t* pre_bf = (bf16_t*)p; p += align_up((size_t)BATCH * CELL * sizeof(bf16_t));
  float*  c_ws   = (float*)p;  p += align_up((size_t)BATCH * CELL * sizeof(float));

  // fp32 -> bf16 conversions (every call; stateless & deterministic).
  f32_to_bf16_kernel<<<(int)((nx + 255) / 256), 256, 0, stream>>>(x, x_bf, nx);
  f32_to_bf16_kernel<<<(int)((nwi + 255) / 256), 256, 0, stream>>>(W_in, Win_bf, nwi);
  f32_to_bf16_kernel<<<(int)((nws + 255) / 256), 256, 0, stream>>>(W_state, Wst_bf, nws);
  f32_to_bf16_kernel<<<(int)((nwp + 255) / 256), 256, 0, stream>>>(W_proj, Wp_bf, nwp);
  init_state_kernel<<<(BATCH * CELL + 255) / 256, 256, 0, stream>>>(h_bf, c_ws);

  for (int t = 0; t < TSTEPS; ++t) {
    lstm_gates_cell_kernel<<<32, 256, 0, stream>>>(x_bf, Win_bf, Wst_bf, b_state,
                                                   h_bf, c_ws, pre_bf, t);
    lstm_proj_kernel<<<32, 256, 0, stream>>>(pre_bf, Wp_bf, out, h_bf, t);
  }
  epilogue_kernel<<<(BATCH * CELL + 255) / 256, 256, 0, stream>>>(out, c_ws, out);
}
